// VanillaLinearAttention_62302795596538
// MI455X (gfx1250) — compile-verified, matrix operand order checked by reading, not run on hardware
//
#include <hip/hip_runtime.h>
#include <hip/hip_bf16.h>
#include <math.h>

// ---------------------------------------------------------------------------
// VanillaLinearAttention on MI455X (gfx1250, wave32, WMMA).
// All WMMA operands that come from global memory are PRE-PACKED into bf16
// wave32 fragment layout, so every operand fetch is 2x global_load_b128.
// ---------------------------------------------------------------------------

typedef __attribute__((ext_vector_type(16))) __bf16 v16bf;
typedef __attribute__((ext_vector_type(8)))  float  v8f;

#define TT 2048   // sequence length
#define DD 512    // d_model == H*DH
#define NH 8      // heads
#define DH 64     // head dim
#define CH 32     // attention chunk length
#define KS 16     // k-steps per 512-deep GEMM (512/32)

__device__ __forceinline__ v8f wmma_bf16(v16bf a, v16bf b, v8f c) {
  return __builtin_amdgcn_wmma_f32_16x16x32_bf16(
      /*neg_a=*/false, a, /*neg_b=*/false, b,
      /*c_mod=*/(short)0, c, /*reuse_a=*/false, /*reuse_b=*/false);
}

// Fragment register layout (wave32): lane&15 -> row/col index, lane>>4 = half,
// vgpr r (of 8) holds K pair at kb = (r>>2)*16 + half*8 + (r&3)*2.

// ---- on-the-fly fragment loaders (LDS-resident operands only) --------------
__device__ __forceinline__ v16bf frag_kc(const float* src, int ld, int r0, int k0) {
  const int lane = threadIdx.x & 31;
  const int m    = lane & 15;
  const int half = lane >> 4;
  v16bf f;
#pragma unroll
  for (int r = 0; r < 8; ++r) {
    const int kb = (r >> 2) * 16 + half * 8 + (r & 3) * 2;
    const float* p = src + (size_t)(r0 + m) * ld + (k0 + kb);
    f[2 * r]     = (__bf16)p[0];
    f[2 * r + 1] = (__bf16)p[1];
  }
  return f;
}

__device__ __forceinline__ v16bf frag_ks(const float* src, int ld, int k0, int c0) {
  const int lane = threadIdx.x & 31;
  const int n    = lane & 15;
  const int half = lane >> 4;
  v16bf f;
#pragma unroll
  for (int r = 0; r < 8; ++r) {
    const int kb = (r >> 2) * 16 + half * 8 + (r & 3) * 2;
    const float* p = src + (size_t)(k0 + kb) * ld + c0 + n;
    f[2 * r]     = (__bf16)p[0];
    f[2 * r + 1] = (__bf16)p[ld];
  }
  return f;
}

// ---- packed fragment loads: one contiguous 32B vector per lane -------------
__device__ __forceinline__ v16bf frag_blk(const __bf16* __restrict__ P, int blk) {
  const int lane = threadIdx.x & 31;
  return *(const v16bf*)(P + ((size_t)blk * 32 + lane) * 16);
}
__device__ __forceinline__ v16bf frag_pk(const __bf16* __restrict__ P,
                                         int tile, int ks) {
  return frag_blk(P, tile * KS + ks);
}

// ---------------------------------------------------------------------------
// Pack kernels: fp32 row-major -> bf16 fragments. One thread per fragment
// slice: 16 strided fp32 reads, one 32B contiguous write.
// ---------------------------------------------------------------------------

// A-operand pack over full 512-wide matrix: tile = row/16, K = 512.
__global__ void vla_pack_a_kernel(const float* __restrict__ M,
                                  __bf16* __restrict__ P, int rows) {
  const int tid = blockIdx.x * blockDim.x + threadIdx.x;
  if (tid >= (rows / 16) * KS * 32) return;
  const int lane = tid & 31;
  const int ks   = (tid >> 5) & (KS - 1);
  const int rt   = tid >> 9;
  const int m = lane & 15, half = lane >> 4;
  const float* src = M + (size_t)(rt * 16 + m) * DD + ks * 32;
  v16bf f;
#pragma unroll
  for (int r = 0; r < 8; ++r) {
    const int kb = (r >> 2) * 16 + half * 8 + (r & 3) * 2;
    f[2 * r]     = (__bf16)src[kb];
    f[2 * r + 1] = (__bf16)src[kb + 1];
  }
  *(v16bf*)(P + (size_t)tid * 16) = f;
}

// B-operand pack of a 512x512 weight: tile = col/16, K = 512.
__global__ void vla_pack_b_kernel(const float* __restrict__ W,
                                  __bf16* __restrict__ P) {
  const int tid = blockIdx.x * blockDim.x + threadIdx.x;  // 32*16*32 = 16384
  const int lane = tid & 31;
  const int ks   = (tid >> 5) & (KS - 1);
  const int ct   = tid >> 9;
  const int n = lane & 15, half = lane >> 4;
  const float* src = W + (size_t)(ks * 32) * DD + ct * 16 + n;
  v16bf f;
#pragma unroll
  for (int r = 0; r < 8; ++r) {
    const int kb = (r >> 2) * 16 + half * 8 + (r & 3) * 2;
    f[2 * r]     = (__bf16)src[(size_t)kb * DD];
    f[2 * r + 1] = (__bf16)src[(size_t)(kb + 1) * DD];
  }
  *(v16bf*)(P + (size_t)tid * 16) = f;
}

// Per-head A-style pack (lane -> time row, K = DH = 64, 2 k-steps).
// blk = (h*128 + time_tile)*2 + ks; tid = (blk*32 + lane).
__global__ void vla_pack_ah_kernel(const float* __restrict__ M,
                                   __bf16* __restrict__ P) {
  const int tid = blockIdx.x * blockDim.x + threadIdx.x;  // 8*128*2*32 = 65536
  const int lane = tid & 31;
  const int ks   = (tid >> 5) & 1;
  const int tt   = (tid >> 6) & 127;
  const int h    = tid >> 13;
  const int m = lane & 15, half = lane >> 4;
  const float* src = M + (size_t)(tt * 16 + m) * DD + h * DH + ks * 32;
  v16bf f;
#pragma unroll
  for (int r = 0; r < 8; ++r) {
    const int kb = (r >> 2) * 16 + half * 8 + (r & 3) * 2;
    f[2 * r]     = (__bf16)src[kb];
    f[2 * r + 1] = (__bf16)src[kb + 1];
  }
  *(v16bf*)(P + (size_t)tid * 16) = f;
}

// Per-head T-style pack (lane -> DH col, K = time-within-chunk = 32).
// blk = (h*64 + chunk)*4 + col_tile; tid = (blk*32 + lane).
__global__ void vla_pack_th_kernel(const float* __restrict__ M,
                                   __bf16* __restrict__ P) {
  const int tid = blockIdx.x * blockDim.x + threadIdx.x;  // 8*64*4*32 = 65536
  const int lane = tid & 31;
  const int ct   = (tid >> 5) & 3;
  const int ck   = (tid >> 7) & 63;
  const int h    = tid >> 13;
  const int n = lane & 15, half = lane >> 4;
  const float* src = M + (size_t)(ck * CH) * DD + h * DH + ct * 16 + n;
  v16bf f;
#pragma unroll
  for (int r = 0; r < 8; ++r) {
    const int kb = (r >> 2) * 16 + half * 8 + (r & 3) * 2;
    f[2 * r]     = (__bf16)src[(size_t)kb * DD];
    f[2 * r + 1] = (__bf16)src[(size_t)(kb + 1) * DD];
  }
  *(v16bf*)(P + (size_t)tid * 16) = f;
}

// ---------------------------------------------------------------------------
// Kernel 1: fused QKV projection + elu_plus_one on Q,K, from packed operands.
// ---------------------------------------------------------------------------
__global__ void vla_qkv_kernel(const __bf16* __restrict__ Xpk,
                               const __bf16* __restrict__ Wqpk, const float* __restrict__ bq,
                               const __bf16* __restrict__ Wkpk, const float* __restrict__ bk,
                               const __bf16* __restrict__ Wvpk, const float* __restrict__ bv,
                               float* __restrict__ Qp, float* __restrict__ Kp,
                               float* __restrict__ Vb) {
  const int lane = threadIdx.x & 31;
  const int wave = threadIdx.x >> 5;
  const int rt   = blockIdx.y;
  const int ct   = blockIdx.x * 8 + wave;       // 0..95
  const int proj = ct >> 5;                     // 0=Q 1=K 2=V
  const int ctl  = ct & 31;
  const int n0   = ctl * 16;

  const __bf16* Wpk  = (proj == 0) ? Wqpk : (proj == 1) ? Wkpk : Wvpk;
  const float*  bias = (proj == 0) ? bq : (proj == 1) ? bk : bv;
  float*        out  = (proj == 0) ? Qp : (proj == 1) ? Kp : Vb;

  v8f acc = {};
#pragma unroll
  for (int ks = 0; ks < KS; ++ks)
    acc = wmma_bf16(frag_pk(Xpk, rt, ks), frag_pk(Wpk, ctl, ks), acc);

  const int n = lane & 15, half = lane >> 4;
  const float bb = bias[n0 + n];
#pragma unroll
  for (int r = 0; r < 8; ++r) {
    float v = acc[r] + bb;
    if (proj < 2) v = (v > 0.f) ? (v + 1.f) : __expf(v);  // elu(x)+1
    out[(size_t)(rt * 16 + r + 8 * half) * DD + n0 + n] = v;
  }
}

// ---------------------------------------------------------------------------
// Kernel 2: chunked causal linear attention, packed global operands.
// One block (8 waves) per head, sequential over T/CH chunks.
// ---------------------------------------------------------------------------
__global__ void vla_attn_kernel(const float* __restrict__ Qp,
                                const float* __restrict__ Kp,
                                const __bf16* __restrict__ Qpa,  // A-style per head
                                const __bf16* __restrict__ Kpa,  // A-style per head
                                const __bf16* __restrict__ Kpt,  // T-style per head
                                const __bf16* __restrict__ Vpt,  // T-style per head
                                float* __restrict__ O) {
  const int h    = blockIdx.x;
  const int lane = threadIdx.x & 31;
  const int wave = threadIdx.x >> 5;
  const int col  = h * DH;

  __shared__ float Ssh[DH * DH];                // running K^T V state, fp32
  __shared__ float zsh[DH];                     // running sum of Kp rows
  __shared__ float sc[CH * CH];                 // intra-chunk scores
  __shared__ float den[CH];

  for (int i = threadIdx.x; i < DH * DH; i += blockDim.x) Ssh[i] = 0.f;
  if (threadIdx.x < DH) zsh[threadIdx.x] = 0.f;
  __syncthreads();

  for (int c = 0; c < TT; c += CH) {
    const int tt0 = c >> 4;                     // first time tile of chunk
    const int ck  = c >> 5;                     // chunk index

    // ---- 1: raw scores = Qp_c (32x64) @ Kp_c^T (64x32), waves 0..3 ----------
    if (wave < 4) {
      const int mt = wave >> 1, nt = wave & 1;
      v8f acc = {};
#pragma unroll
      for (int ks = 0; ks < 2; ++ks) {
        v16bf a = frag_blk(Qpa, (h * 128 + tt0 + mt) * 2 + ks);
        v16bf b = frag_blk(Kpa, (h * 128 + tt0 + nt) * 2 + ks);  // Kp^T view
        acc = wmma_bf16(a, b, acc);
      }
      const int n = lane & 15, half = lane >> 4;
#pragma unroll
      for (int r = 0; r < 8; ++r)
        sc[(mt * 16 + r + 8 * half) * CH + nt * 16 + n] = acc[r];
    }
    __syncthreads();

    // ---- 2a: causal mask (zero j > i) --------------------------------------
    for (int idx = threadIdx.x; idx < CH * CH; idx += blockDim.x) {
      const int i = idx >> 5, j = idx & 31;
      if (j > i) sc[idx] = 0.f;
    }
    __syncthreads();

    // ---- 2b: denominator ---------------------------------------------------
    if (threadIdx.x < CH) {
      const int t = threadIdx.x;
      float d = 0.f;
      const float* q = Qp + (size_t)(c + t) * DD + col;
      for (int k = 0; k < DH; ++k) d += q[k] * zsh[k];
      for (int j = 0; j <= t; ++j) d += sc[t * CH + j];
      den[t] = fmaxf(d, 1e-6f);
    }
    __syncthreads();

    // ---- 2c: z += column-sum of Kp chunk ----------------------------------
    if (threadIdx.x < DH) {
      const int k = threadIdx.x;
      float s = zsh[k];
      for (int j = 0; j < CH; ++j) s += Kp[(size_t)(c + j) * DD + col + k];
      zsh[k] = s;
    }

    // ---- 3: out = (Qp_c @ S_prev + masked_scores @ V_c) / den --------------
    {
      const int mt = wave >> 2, nt = wave & 3;
      v8f acc = {};
#pragma unroll
      for (int ks = 0; ks < 2; ++ks) {
        v16bf a = frag_blk(Qpa, (h * 128 + tt0 + mt) * 2 + ks);
        v16bf b = frag_ks(Ssh, DH, ks * 32, nt * 16);
        acc = wmma_bf16(a, b, acc);
      }
      {
        v16bf a = frag_kc(sc, CH, mt * 16, 0);
        v16bf b = frag_blk(Vpt, (h * 64 + ck) * 4 + nt);
        acc = wmma_bf16(a, b, acc);
      }
      const int n = lane & 15, half = lane >> 4;
#pragma unroll
      for (int r = 0; r < 8; ++r) {
        const int t = mt * 16 + r + 8 * half;
        O[(size_t)(c + t) * DD + col + nt * 16 + n] = acc[r] / den[t];
      }
    }
    __syncthreads();   // all reads of S_prev complete before update

    // ---- 4: S += Kp_c^T (64x32) @ V_c (32x64); 16 tiles, 2 per wave --------
#pragma unroll
    for (int tt = 0; tt < 2; ++tt) {
      const int tile = wave * 2 + tt;
      const int mt = tile >> 2, nt = tile & 3;
      const int n = lane & 15, half = lane >> 4;
      v8f cf;
#pragma unroll
      for (int r = 0; r < 8; ++r)
        cf[r] = Ssh[(mt * 16 + r + 8 * half) * DH + nt * 16 + n];
      v16bf a = frag_blk(Kpt, (h * 64 + ck) * 4 + mt);
      v16bf b = frag_blk(Vpt, (h * 64 + ck) * 4 + nt);
      cf = wmma_bf16(a, b, cf);
#pragma unroll
      for (int r = 0; r < 8; ++r)
        Ssh[(mt * 16 + r + 8 * half) * DH + nt * 16 + n] = cf[r];
    }
    __syncthreads();
  }
}

// ---------------------------------------------------------------------------
// Kernel 3: output projection out = attn @ Wo + bo, from packed operands.
// ---------------------------------------------------------------------------
__global__ void vla_oproj_kernel(const __bf16* __restrict__ Apk,
                                 const __bf16* __restrict__ Wopk,
                                 const float* __restrict__ bo,
                                 float* __restrict__ out) {
  const int lane = threadIdx.x & 31;
  const int wave = threadIdx.x >> 5;
  const int rt   = blockIdx.y;
  const int ctl  = blockIdx.x * 8 + wave;
  const int n0   = ctl * 16;

  v8f acc = {};
#pragma unroll
  for (int ks = 0; ks < KS; ++ks)
    acc = wmma_bf16(frag_pk(Apk, rt, ks), frag_pk(Wopk, ctl, ks), acc);

  const int n = lane & 15, half = lane >> 4;
  const float bb = bo[n0 + n];
#pragma unroll
  for (int r = 0; r < 8; ++r)
    out[(size_t)(rt * 16 + r + 8 * half) * DD + n0 + n] = acc[r] + bb;
}

// ---------------------------------------------------------------------------
extern "C" void kernel_launch(void* const* d_in, const int* in_sizes, int n_in,
                              void* d_out, int out_size, void* d_ws, size_t ws_size,
                              hipStream_t stream) {
  const float* x  = (const float*)d_in[0];
  const float* Wq = (const float*)d_in[1];
  const float* bq = (const float*)d_in[2];
  const float* Wk = (const float*)d_in[3];
  const float* bk = (const float*)d_in[4];
  const float* Wv = (const float*)d_in[5];
  const float* bv = (const float*)d_in[6];
  const float* Wo = (const float*)d_in[7];
  const float* bo = (const float*)d_in[8];
  float* out = (float*)d_out;

  const size_t mat  = (size_t)TT * DD;     // 1M elems
  const size_t wmat = (size_t)DD * DD;     // 256K elems
  float* Qp   = (float*)d_ws;
  float* Kp   = Qp + mat;
  float* Vb   = Kp + mat;
  float* attn = Vb + mat;
  __bf16* Xpk    = (__bf16*)(attn + mat);
  __bf16* Wqpk   = Xpk + mat;
  __bf16* Wkpk   = Wqpk + wmat;
  __bf16* Wvpk   = Wkpk + wmat;
  __bf16* Wopk   = Wvpk + wmat;
  __bf16* attnpk = Wopk + wmat;
  __bf16* Qpa    = attnpk + mat;
  __bf16* Kpa    = Qpa + mat;
  __bf16* Kpt    = Kpa + mat;
  __bf16* Vpt    = Kpt + mat;

  // 0) pack x and weights into bf16 WMMA-fragment layout
  vla_pack_a_kernel<<<dim3(256), 256, 0, stream>>>(x, Xpk, TT);
  vla_pack_b_kernel<<<dim3(64), 256, 0, stream>>>(Wq, Wqpk);
  vla_pack_b_kernel<<<dim3(64), 256, 0, stream>>>(Wk, Wkpk);
  vla_pack_b_kernel<<<dim3(64), 256, 0, stream>>>(Wv, Wvpk);
  vla_pack_b_kernel<<<dim3(64), 256, 0, stream>>>(Wo, Wopk);

  // 1) QKV projections + elu_plus_one (Q,K)
  vla_qkv_kernel<<<dim3(12, 128), 256, 0, stream>>>(Xpk, Wqpk, bq, Wkpk, bk,
                                                    Wvpk, bv, Qp, Kp, Vb);

  // 1b) pack attention operands per head (A-style and T-style fragments)
  vla_pack_ah_kernel<<<dim3(256), 256, 0, stream>>>(Qp, Qpa);
  vla_pack_ah_kernel<<<dim3(256), 256, 0, stream>>>(Kp, Kpa);
  vla_pack_th_kernel<<<dim3(256), 256, 0, stream>>>(Kp, Kpt);
  vla_pack_th_kernel<<<dim3(256), 256, 0, stream>>>(Vb, Vpt);

  // 2) causal chunked linear attention: one workgroup per head
  vla_attn_kernel<<<dim3(NH), 256, 0, stream>>>(Qp, Kp, Qpa, Kpa, Kpt, Vpt, attn);

  // 3) pack attention output, then output projection
  vla_pack_a_kernel<<<dim3(256), 256, 0, stream>>>(attn, attnpk, TT);
  vla_oproj_kernel<<<dim3(4, 128), 256, 0, stream>>>(attnpk, Wopk, bo, out);
}
